// CausalSelfAttention_72533407695457
// MI455X (gfx1250) — matrix operand order repaired and checked by reading, not compiled, not run on hardware
//
#include <hip/hip_runtime.h>
#include <hip/hip_bf16.h>
#include <math.h>

// Problem constants (match reference)
#define BB     4
#define TT     2048
#define CC     2048
#define NHEAD  16
#define NKV    4
#define HD     128
#define KVD    512
#define WIN    256

typedef __attribute__((ext_vector_type(16))) _Float16 v16h;
typedef __attribute__((ext_vector_type(8)))  float    v8f;

static __device__ __forceinline__ v8f wmma_f16(v16h a, v16h b, v8f c) {
  // D = A(16x32 f16) * B(32x16 f16) + C(16x16 f32)
  return __builtin_amdgcn_wmma_f32_16x16x32_f16(false, a, false, b, (short)0, c,
                                                false, false);
}

// Async global->LDS copy, 16 bytes per lane (ASYNCcnt-tracked).
// lds_off = LDS byte offset (low 32 bits of flat shared address).
static __device__ __forceinline__ void async_cp16(unsigned lds_off,
                                                  const void* gaddr) {
  asm volatile("global_load_async_to_lds_b128 %0, %1, off" ::"v"(lds_off),
               "v"(gaddr)
               : "memory");
}
static __device__ __forceinline__ void async_cp16_o16(unsigned lds_off,
                                                      const void* gaddr) {
  // INST_OFFSET applies to both the global and LDS address (ISA 08 §4.4)
  asm volatile("global_load_async_to_lds_b128 %0, %1, off offset:16" ::"v"(
                   lds_off),
               "v"(gaddr)
               : "memory");
}

// ---------------------------------------------------------------------------
// f32 -> f16 bulk convert (n % 256 == 0 for all uses)
// ---------------------------------------------------------------------------
__global__ void f32_to_f16(const float* __restrict__ in,
                           _Float16* __restrict__ out, int n) {
  const int idx = blockIdx.x * 256 + threadIdx.x;
  if (idx < n) out[idx] = (_Float16)in[idx];
}

// ---------------------------------------------------------------------------
// Tiled WMMA GEMM, all-f16 inputs: C = A(f16) * B(f16).
// Block tile 128x128, K-step 32, double-buffered LDS filled by
// global_load_async_to_lds_b128. 256 threads = 8 waves; each wave owns a
// 64x32 sub-tile = 4x2 WMMA accumulators.
// Requires M%128==0, N%128==0, K%32==0.
// ---------------------------------------------------------------------------
template <bool WRITE_HALF>
__global__ __launch_bounds__(256)
void gemm_async_128x128(const _Float16* __restrict__ A,
                        const _Float16* __restrict__ B, void* __restrict__ Cv,
                        int M, int N, int K) {
  __shared__ _Float16 As[2][128][40];   // 128 rows x 32 k (+8 pad), x2 bufs
  __shared__ _Float16 Bs[2][32][136];   // 32 k x 128 cols (+8 pad), x2 bufs

  const int tid  = threadIdx.x;
  const int lane = tid & 31;
  const int wave = tid >> 5;
  const int m0 = blockIdx.y * 128;
  const int n0 = blockIdx.x * 128;
  const int wm = (wave >> 2) * 64;
  const int wn = (wave & 3) * 32;

  const int nidx = lane & 15;
  const int grpA = (lane < 16) ? 0 : 16;   // A-fragment k-half base
  const int grpB = (lane < 16) ? 0 : 16;   // B-fragment k base
  const int grpC = (lane < 16) ? 0 : 8;    // C/D row-group offset

  v8f acc[4][2];
  #pragma unroll
  for (int mt = 0; mt < 4; ++mt)
    #pragma unroll
    for (int nt = 0; nt < 2; ++nt)
      #pragma unroll
      for (int j = 0; j < 8; ++j) acc[mt][nt][j] = 0.0f;

  // staging assignment: 16 halves (32 B) per thread per tile
  const int arow = tid >> 1;          // 128 rows, 2 threads/row
  const int acol = (tid & 1) * 16;
  const int brow = tid >> 3;          // 32 rows, 8 threads/row
  const int bcol = (tid & 7) * 16;

  const _Float16* Ag = A + (size_t)(m0 + arow) * K + acol;
  const _Float16* Bg = B + (size_t)brow * N + (n0 + bcol);

  // issue one chunk's async copies (4 x 16B per thread)
  auto issue = [&](int k0, int buf) {
    const unsigned a_lds = (unsigned)(uintptr_t)&As[buf][arow][acol];
    const unsigned b_lds = (unsigned)(uintptr_t)&Bs[buf][brow][bcol];
    const _Float16* ag = Ag + k0;
    const _Float16* bg = Bg + (size_t)k0 * N;
    async_cp16(a_lds, ag);
    async_cp16_o16(a_lds, ag);
    async_cp16(b_lds, bg);
    async_cp16_o16(b_lds, bg);
  };

  issue(0, 0);

  int buf = 0;
  for (int k0 = 0; k0 < K; k0 += 32) {
    const bool has_next = (k0 + 32) < K;
    if (has_next) {
      issue(k0 + 32, buf ^ 1);
      asm volatile("s_wait_asynccnt 0x4" ::: "memory");
    } else {
      asm volatile("s_wait_asynccnt 0x0" ::: "memory");
    }
    __syncthreads();   // all threads' copies into As/Bs[buf] visible

    v16h af[4];
    v16h bf[2];
    #pragma unroll
    for (int mt = 0; mt < 4; ++mt) {
      #pragma unroll
      for (int i = 0; i < 8; ++i) {
        af[mt][i]     = As[buf][wm + mt * 16 + nidx][grpA + i];
        af[mt][i + 8] = As[buf][wm + mt * 16 + nidx][grpA + 8 + i];
      }
    }
    #pragma unroll
    for (int nt = 0; nt < 2; ++nt) {
      #pragma unroll
      for (int i = 0; i < 16; ++i)
        bf[nt][i] = Bs[buf][grpB + i][wn + nt * 16 + nidx];
    }

    #pragma unroll
    for (int mt = 0; mt < 4; ++mt)
      #pragma unroll
      for (int nt = 0; nt < 2; ++nt)
        acc[mt][nt] = wmma_f16(af[mt], bf[nt], acc[mt][nt]);

    __syncthreads();   // done reading buf before it is refilled next iter
    buf ^= 1;
  }

  #pragma unroll
  for (int mt = 0; mt < 4; ++mt)
    #pragma unroll
    for (int nt = 0; nt < 2; ++nt)
      #pragma unroll
      for (int v = 0; v < 8; ++v) {
        const size_t idx = (size_t)(m0 + wm + mt * 16 + v + grpC) * N +
                           (n0 + wn + nt * 16 + nidx);
        if (WRITE_HALF)
          ((_Float16*)Cv)[idx] = (_Float16)acc[mt][nt][v];
        else
          ((float*)Cv)[idx] = acc[mt][nt][v];
      }
}

// ---------------------------------------------------------------------------
// RoPE on f16 data. grid = (B*T, n_head), block = 64.
// ---------------------------------------------------------------------------
__global__ void rope_f16(const _Float16* __restrict__ in,
                         _Float16* __restrict__ out, int n_head) {
  const int row = blockIdx.x;              // b*T + t
  const int h   = blockIdx.y;
  const int j   = threadIdx.x;             // 0..63
  const int t   = row & (TT - 1);
  const int dim = n_head * HD;
  // inv_freq = 10000^(-j/64) = exp(-j * ln(1e4)/64)
  const float inv = __expf(-(float)j * 0.14391156511f);
  const float ang = (float)t * inv;
  float s, c;
  __sincosf(ang, &s, &c);
  const size_t base = (size_t)row * dim + (size_t)h * HD;
  const float x1 = (float)in[base + j];
  const float x2 = (float)in[base + 64 + j];
  out[base + j]      = (_Float16)(x1 * c - x2 * s);
  out[base + 64 + j] = (_Float16)(x1 * s + x2 * c);
}

// ---------------------------------------------------------------------------
// V transpose: (B*T, 512) f16 -> [b][kv][d][T] f16 (contiguous in T).
// ---------------------------------------------------------------------------
__global__ void v_transpose(const _Float16* __restrict__ in,
                            _Float16* __restrict__ out) {
  const size_t idx = (size_t)blockIdx.x * blockDim.x + threadIdx.x;
  const size_t total = (size_t)BB * TT * KVD;
  if (idx >= total) return;
  const int col = (int)(idx % KVD);
  const int row = (int)(idx / KVD);
  const int b = row / TT, t = row & (TT - 1);
  const int kv = col / HD, d = col & (HD - 1);
  out[((size_t)(b * NKV + kv) * HD + d) * TT + t] = in[idx];
}

// ---------------------------------------------------------------------------
// Flash attention with sliding-window causal mask.
// grid = (T/128, NHEAD, B), block = 256 (8 waves). Wave w owns queries
// [q0, q0+15]; keys walked in chunks of 32 over [max(0,q0-WIN) & ~31, q0+15].
// ---------------------------------------------------------------------------
__global__ __launch_bounds__(256)
void attn_kernel(const _Float16* __restrict__ Qh,
                 const _Float16* __restrict__ Kh,
                 const _Float16* __restrict__ Vt,
                 _Float16* __restrict__ Yh) {
  __shared__ _Float16 Plds[8][16][32];     // per-wave P round-trip scratch

  const int tid  = threadIdx.x;
  const int lane = tid & 31;
  const int wave = tid >> 5;
  const int b = blockIdx.z;
  const int h = blockIdx.y;
  const int q0  = blockIdx.x * 128 + wave * 16;
  const int kvh = h / (NHEAD / NKV);

  const int nidx = lane & 15;
  const int grp8 = (lane < 16) ? 0 : 8;    // C/D row group
  const int grpA = (lane < 16) ? 0 : 16;   // A-fragment k-half base
  const int grpB = (lane < 16) ? 0 : 16;   // B-fragment k base

  // Q fragments: 16x128 split into 4 K=32 chunks
  v16h qf[4];
  {
    const size_t qbase =
        (size_t)(b * TT + q0 + nidx) * CC + (size_t)h * HD;
    #pragma unroll
    for (int dc = 0; dc < 4; ++dc) {
      #pragma unroll
      for (int i = 0; i < 8; ++i) {
        qf[dc][i]     = Qh[qbase + dc * 32 + grpA + i];
        qf[dc][i + 8] = Qh[qbase + dc * 32 + grpA + 8 + i];
      }
    }
  }

  v8f acc[8];                              // 16x128 output, 8 d-tiles
  #pragma unroll
  for (int dt = 0; dt < 8; ++dt)
    #pragma unroll
    for (int j = 0; j < 8; ++j) acc[dt][j] = 0.0f;

  float mrow[8], lrow[8];
  #pragma unroll
  for (int v = 0; v < 8; ++v) { mrow[v] = -INFINITY; lrow[v] = 0.0f; }

  int lo = q0 - WIN;
  if (lo < 0) lo = 0;
  const int kb0 = lo & ~31;
  const int hi  = q0 + 15;
  const float scale = 0.08838834764831845f;   // 1/sqrt(128)

  for (int kb = kb0; kb <= hi; kb += 32) {
    // S = Q K^T for 32 keys -> two 16x16 f32 tiles
    v8f s0, s1;
    #pragma unroll
    for (int j = 0; j < 8; ++j) { s0[j] = 0.0f; s1[j] = 0.0f; }

    #pragma unroll
    for (int dc = 0; dc < 4; ++dc) {
      v16h kf0, kf1;
      const size_t kbase0 = (size_t)(b * TT + kb + nidx) * KVD +
                            (size_t)kvh * HD + dc * 32 + grpB;
      const size_t kbase1 = kbase0 + (size_t)16 * KVD;
      #pragma unroll
      for (int i = 0; i < 16; ++i) {
        kf0[i] = Kh[kbase0 + i];
        kf1[i] = Kh[kbase1 + i];
      }
      s0 = wmma_f16(qf[dc], kf0, s0);
      s1 = wmma_f16(qf[dc], kf1, s1);
    }

    // mask + online softmax (per-row stats live in 16-lane groups)
    #pragma unroll
    for (int v = 0; v < 8; ++v) {
      const int qi  = q0 + v + grp8;
      const int k0i = kb + nidx;
      const int k1i = kb + 16 + nidx;
      float x0 = (k0i <= qi && k0i >= qi - WIN) ? s0[v] * scale : -INFINITY;
      float x1 = (k1i <= qi && k1i >= qi - WIN) ? s1[v] * scale : -INFINITY;
      float rm = fmaxf(x0, x1);
      #pragma unroll
      for (int off = 1; off < 16; off <<= 1)
        rm = fmaxf(rm, __shfl_xor(rm, off, 32));
      const float mnew  = fmaxf(mrow[v], rm);
      const float alpha = __expf(mrow[v] - mnew);
      const float p0 = __expf(x0 - mnew);
      const float p1 = __expf(x1 - mnew);
      float rs = p0 + p1;
      #pragma unroll
      for (int off = 1; off < 16; off <<= 1)
        rs += __shfl_xor(rs, off, 32);
      lrow[v] = lrow[v] * alpha + rs;
      mrow[v] = mnew;
      #pragma unroll
      for (int dt = 0; dt < 8; ++dt) acc[dt][v] *= alpha;
      // stash P in C/D layout (per-wave private LDS region)
      Plds[wave][v + grp8][nidx]      = (_Float16)p0;
      Plds[wave][v + grp8][16 + nidx] = (_Float16)p1;
    }

    // per-wave DS ordering: writes above must land before A-layout reads
    asm volatile("s_wait_dscnt 0" ::: "memory");

    v16h pf;
    #pragma unroll
    for (int i = 0; i < 8; ++i) {
      pf[i]     = Plds[wave][nidx][grpA + i];
      pf[i + 8] = Plds[wave][nidx][grpA + 8 + i];
    }

    // O += P (16x32) * V (32x128); Vt is contiguous along T
    #pragma unroll
    for (int dt = 0; dt < 8; ++dt) {
      v16h vf;
      const size_t vb =
          ((size_t)(b * NKV + kvh) * HD + dt * 16 + nidx) * TT + kb + grpB;
      #pragma unroll
      for (int i = 0; i < 16; ++i) vf[i] = Vt[vb + i];
      acc[dt] = wmma_f16(pf, vf, acc[dt]);
    }
  }

  // normalize and store (B*T, 2048) f16
  #pragma unroll
  for (int dt = 0; dt < 8; ++dt)
    #pragma unroll
    for (int v = 0; v < 8; ++v) {
      const int qi = q0 + v + grp8;
      Yh[(size_t)(b * TT + qi) * CC + (size_t)h * HD + dt * 16 + nidx] =
          (_Float16)(acc[dt][v] / lrow[v]);
    }
}

// ---------------------------------------------------------------------------
// Host launcher
// ---------------------------------------------------------------------------
extern "C" void kernel_launch(void* const* d_in, const int* in_sizes, int n_in,
                              void* d_out, int out_size, void* d_ws,
                              size_t ws_size, hipStream_t stream) {
  (void)in_sizes; (void)n_in; (void)out_size; (void)ws_size;
  const float* x  = (const float*)d_in[0];
  const float* Wq = (const float*)d_in[1];
  const float* Wk = (const float*)d_in[2];
  const float* Wv = (const float*)d_in[3];
  const float* Wo = (const float*)d_in[4];
  float* out = (float*)d_out;
  char* ws = (char*)d_ws;

  const int MQ = BB * TT;  // 8192

  // Workspace layout (148 MB total):
  _Float16* xh   = (_Float16*)(ws + ((size_t)0 << 20));    // 32 MB
  _Float16* Wqh  = (_Float16*)(ws + ((size_t)32 << 20));   //  8 MB
  _Float16* Wkh  = (_Float16*)(ws + ((size_t)40 << 20));   //  2 MB
  _Float16* Wvh  = (_Float16*)(ws + ((size_t)42 << 20));   //  2 MB
  _Float16* Woh  = (_Float16*)(ws + ((size_t)44 << 20));   //  8 MB
  _Float16* Qraw = (_Float16*)(ws + ((size_t)52 << 20));   // 32 MB (pre-rope)
  _Float16* Kraw = (_Float16*)(ws + ((size_t)84 << 20));   //  8 MB
  _Float16* Vraw = (_Float16*)(ws + ((size_t)92 << 20));   //  8 MB
  _Float16* Qh   = (_Float16*)(ws + ((size_t)100 << 20));  // 32 MB (post-rope)
  _Float16* Kh   = (_Float16*)(ws + ((size_t)132 << 20));  //  8 MB
  _Float16* Vt   = (_Float16*)(ws + ((size_t)140 << 20));  //  8 MB
  _Float16* Yh   = (_Float16*)(ws + ((size_t)52 << 20));   // reuse Qraw (dead)

  dim3 blk(256);

  // f32 -> f16 pre-pass (halves GEMM read traffic; enables async LDS staging)
  f32_to_f16<<<(MQ * CC) / 256, blk, 0, stream>>>(x, xh, MQ * CC);
  f32_to_f16<<<(CC * CC) / 256, blk, 0, stream>>>(Wq, Wqh, CC * CC);
  f32_to_f16<<<(CC * KVD) / 256, blk, 0, stream>>>(Wk, Wkh, CC * KVD);
  f32_to_f16<<<(CC * KVD) / 256, blk, 0, stream>>>(Wv, Wvh, CC * KVD);
  f32_to_f16<<<(CC * CC) / 256, blk, 0, stream>>>(Wo, Woh, CC * CC);

  // QKV projections (async-LDS staged, f16 out)
  gemm_async_128x128<true><<<dim3(CC / 128, MQ / 128), blk, 0, stream>>>(
      xh, Wqh, Qraw, MQ, CC, CC);
  gemm_async_128x128<true><<<dim3(KVD / 128, MQ / 128), blk, 0, stream>>>(
      xh, Wkh, Kraw, MQ, KVD, CC);
  gemm_async_128x128<true><<<dim3(KVD / 128, MQ / 128), blk, 0, stream>>>(
      xh, Wvh, Vraw, MQ, KVD, CC);

  // RoPE and V transpose
  rope_f16<<<dim3(MQ, NHEAD), 64, 0, stream>>>(Qraw, Qh, NHEAD);
  rope_f16<<<dim3(MQ, NKV), 64, 0, stream>>>(Kraw, Kh, NKV);
  v_transpose<<<((size_t)BB * TT * KVD + 255) / 256, 256, 0, stream>>>(Vraw,
                                                                       Vt);

  // Flash attention
  attn_kernel<<<dim3(TT / 128, NHEAD, BB), 256, 0, stream>>>(Qh, Kh, Vt, Yh);

  // Output projection (f32 out)
  gemm_async_128x128<false><<<dim3(CC / 128, MQ / 128), blk, 0, stream>>>(
      Yh, Woh, out, MQ, CC, CC);
}
